// LFQ_51831665328666
// MI455X (gfx1250) — compile-verified, matrix operand order checked by reading, not run on hardware
//
#include <hip/hip_runtime.h>
#include <cstdint>

typedef __attribute__((ext_vector_type(16))) _Float16 v16h;
typedef __attribute__((ext_vector_type(8)))  float    v8f;
typedef __attribute__((ext_vector_type(4)))  float    v4f;

#define DIM      14
#define CS       16384            // 2^14 codes
#define BN       8192             // 4*2048 samples
#define INVT2    200.0f           // 2 * inv_temperature
#define LOG_EPS  (-11.512925464970229f)   // log(1e-5)

// ---------------------------------------------------------------------------
// Workspace layout:
//   apack  [512 rowtiles][32 lanes][16 halves]  = 512 KB  (A operand, WMMA lane
//          layout: e<8 -> hi(h[dim e+8*khalf]), e>=8 -> lo of same dim;
//          dims >= 14 are zero)
//   rowc   [BN] f32   rowconst = 200*sum|h| + sum log1p(exp(-400|h|)) (= lse)
//   colsum [CS] f32   sum over samples of p(code j)
//   scal   [4]  f32   [0]=sum p*max(lp,LOG_EPS) [1]=commit sum [2]=codebook ent
// ---------------------------------------------------------------------------

__global__ void lfq_zero(float* __restrict__ colsum, float* __restrict__ scal) {
    int i = blockIdx.x * blockDim.x + threadIdx.x;
    if (i < CS) colsum[i] = 0.0f;
    if (i < 4)  scal[i]   = 0.0f;
}

// One thread per sample row: h = x@W_in.T + b_in, quantize, indices,
// out = sign(h)@W_out.T + b_out, commit loss, row constants, and the A-operand
// packed directly into the per-lane WMMA layout (hi/lo f16 split).
__global__ void lfq_prologue(const float* __restrict__ x,
                             const float* __restrict__ W_in,
                             const float* __restrict__ b_in,
                             const float* __restrict__ W_out,
                             const float* __restrict__ b_out,
                             _Float16* __restrict__ apack,
                             float* __restrict__ rowc,
                             float* __restrict__ out,
                             float* __restrict__ idxout,
                             float* __restrict__ scal) {
    int n = blockIdx.x * blockDim.x + threadIdx.x;
    if (n >= BN) return;

    float xr[DIM];
#pragma unroll
    for (int k = 0; k < DIM; ++k) xr[k] = x[n * DIM + k];

    float h[16], q[DIM];
    float abss = 0.0f, logz = 0.0f, commit = 0.0f;
    int idx = 0;
#pragma unroll
    for (int d = 0; d < DIM; ++d) {
        float acc = b_in[d];
#pragma unroll
        for (int k = 0; k < DIM; ++k) acc = fmaf(xr[k], W_in[d * DIM + k], acc);
        h[d] = acc;
        float a = fabsf(acc);
        abss += a;
        logz += log1pf(__expf(-2.0f * INVT2 * a));
        float qq = acc > 0.0f ? 1.0f : -1.0f;
        q[d] = qq;
        idx |= (acc > 0.0f ? 1 : 0) << (DIM - 1 - d);   // MSB-first bit mask
        float dd = acc - qq;
        commit += dd * dd;
    }
    h[14] = 0.0f;
    h[15] = 0.0f;

    rowc[n] = INVT2 * abss + logz;

    // Pack A operand: row n -> lane (n&15) gets dims 0..7, lane (n&15)+16 gets
    // dims 8..15 (14,15 zero). Element e<8 = hi half, e>=8 = lo half.
    {
        int rt = n >> 4, l0 = n & 15;
        v16h a0, a1;
#pragma unroll
        for (int e = 0; e < 8; ++e) {
            float v0 = h[e];
            _Float16 hi0 = (_Float16)v0;
            a0[e]     = hi0;
            a0[e + 8] = (_Float16)(v0 - (float)hi0);
            float v1 = h[8 + e];
            _Float16 hi1 = (_Float16)v1;
            a1[e]     = hi1;
            a1[e + 8] = (_Float16)(v1 - (float)hi1);
        }
        *(v16h*)(apack + ((size_t)rt * 32 + l0) * 16)      = a0;
        *(v16h*)(apack + ((size_t)rt * 32 + l0 + 16) * 16) = a1;
    }

#pragma unroll
    for (int dp = 0; dp < DIM; ++dp) {
        float acc = b_out[dp];
#pragma unroll
        for (int d = 0; d < DIM; ++d) acc = fmaf(q[d], W_out[dp * DIM + d], acc);
        out[n * DIM + dp] = acc;
    }
    idxout[n] = (float)idx;

    // commitment loss partial: wave reduce, one atomic per wave
#pragma unroll
    for (int off = 16; off > 0; off >>= 1) commit += __shfl_xor(commit, off, 32);
    if ((threadIdx.x & 31) == 0) atomicAdd(scal + 1, commit);
}

// Heavy kernel: WMMA GEMM h[8192x14] x codes[14x16384] with fused softmax /
// entropy. Each wave owns ONE 16-column code tile (B built once from bit math,
// replicated in both K halves to multiply hi+lo) and loops over 64 row tiles.
// A is preconverted, so the loop is loads + WMMA + exp epilogue only.
__global__ void __launch_bounds__(128) lfq_entropy(const _Float16* __restrict__ apack,
                                                   const float* __restrict__ rowc,
                                                   float* __restrict__ colsum,
                                                   float* __restrict__ scal) {
    const int lane    = threadIdx.x & 31;
    const int wave    = threadIdx.x >> 5;
    const int coltile = blockIdx.x * 4 + wave;      // 0..1023
    const int j       = coltile * 16 + (lane & 15); // this lane's code column (N)
    const int khalf   = lane >> 4;                  // 0: K 0..7/16..23, 1: K 8..15/24..31

    // B matrix (32x16 f16, K x N): B[k][n] = sign of dim (k mod 16) of code n,
    // replicated in both K halves; zero for d >= 14.
    v16h b;
#pragma unroll
    for (int e = 0; e < 8; ++e) {
        int d = e + 8 * khalf;
        _Float16 s = (_Float16)0.0f;
        if (d < DIM)
            s = ((j >> (DIM - 1 - d)) & 1) ? (_Float16)1.0f : (_Float16)(-1.0f);
        b[e]     = s;
        b[e + 8] = s;
    }

    float negent = 0.0f;   // sum of p * max(log p, log eps)
    float csum   = 0.0f;   // sum of p over this lane's rows for column j
    const int t0 = blockIdx.y * 64;
    const v16h* ap = (const v16h*)apack;

#pragma unroll 2
    for (int t = 0; t < 64; ++t) {
        const int rowtile = t0 + t;
        // A operand: preconverted, per-lane layout (2 x b128 load)
        v16h a = ap[(size_t)rowtile * 32 + lane];

        // rowconst for this lane's 8 output rows (m = v + 8*khalf)
        const float* crow = rowc + rowtile * 16 + 8 * khalf;
        v4f c0 = *(const v4f*)(crow);
        v4f c1 = *(const v4f*)(crow + 4);
        float cv[8] = {c0[0], c0[1], c0[2], c0[3], c1[0], c1[1], c1[2], c1[3]};

        v8f cz = {};
        v8f dacc = __builtin_amdgcn_wmma_f32_16x16x32_f16(
            false, a, false, b, (short)0, cz, false, false);

#pragma unroll
        for (int v = 0; v < 8; ++v) {
            float lp = fmaf(INVT2, dacc[v], -cv[v]);  // log p (exact lse), <= 0
            float p  = __expf(lp);
            negent   = fmaf(p, fmaxf(lp, LOG_EPS), negent);
            csum    += p;
        }
    }

    // column sum: lanes l and l^16 hold rows 0..7 and 8..15 of each tile
    float csum2 = csum + __shfl_xor(csum, 16, 32);
    if (lane < 16) atomicAdd(&colsum[j], csum2);

    // entropy: wave reduce -> LDS -> one atomic per block (avoid same-address
    // atomic serialization across 8192 waves)
#pragma unroll
    for (int off = 16; off > 0; off >>= 1) negent += __shfl_xor(negent, off, 32);
    __shared__ float red[4];
    if (lane == 0) red[wave] = negent;
    __syncthreads();
    if (threadIdx.x == 0)
        atomicAdd(scal + 0, red[0] + red[1] + red[2] + red[3]);
}

// Codebook entropy from column sums: avg = S_j / BN; -avg*log(max(avg,eps))
__global__ void lfq_colentropy(const float* __restrict__ colsum,
                               float* __restrict__ scal) {
    int jj = blockIdx.x * blockDim.x + threadIdx.x;
    float term = 0.0f;
    if (jj < CS) {
        float avg = colsum[jj] * (1.0f / (float)BN);
        term = -avg * logf(fmaxf(avg, 1e-5f));
    }
#pragma unroll
    for (int off = 16; off > 0; off >>= 1) term += __shfl_xor(term, off, 32);
    if ((threadIdx.x & 31) == 0) atomicAdd(scal + 2, term);
}

__global__ void lfq_finalize(const float* __restrict__ scal,
                             float* __restrict__ auxout) {
    if (threadIdx.x == 0 && blockIdx.x == 0) {
        float per_sample = -scal[0] / (float)BN;
        float codebook   = scal[2];
        float commit     = scal[1] / (float)(BN * DIM);
        auxout[0] = 0.1f * (per_sample - codebook) + 0.25f * commit;
    }
}

extern "C" void kernel_launch(void* const* d_in, const int* in_sizes, int n_in,
                              void* d_out, int out_size, void* d_ws, size_t ws_size,
                              hipStream_t stream) {
    (void)in_sizes; (void)n_in; (void)out_size; (void)ws_size;
    const float* x     = (const float*)d_in[0];
    const float* W_in  = (const float*)d_in[1];
    const float* b_in  = (const float*)d_in[2];
    const float* W_out = (const float*)d_in[3];
    const float* b_out = (const float*)d_in[4];

    float* out    = (float*)d_out;        // [BN*DIM] f32
    float* idxout = out + BN * DIM;       // [BN] indices (as f32 values)
    float* auxout = idxout + BN;          // [1] aux loss

    _Float16* apack = (_Float16*)d_ws;                    // 512 rowtiles * 32 * 16 halves
    float* rowc   = (float*)d_ws + BN * 16 / 2 * 1 + BN * 8; // = 131072 floats offset
    // compute offsets explicitly in floats: apack occupies 262144 halves = 131072 floats
    rowc          = (float*)d_ws + 131072;
    float* colsum = rowc + BN;
    float* scal   = colsum + CS;

    lfq_zero<<<dim3(CS / 256), dim3(256), 0, stream>>>(colsum, scal);
    lfq_prologue<<<dim3(BN / 256), dim3(256), 0, stream>>>(
        x, W_in, b_in, W_out, b_out, apack, rowc, out, idxout, scal);
    // 256 blocks x 4 waves = 1024 column tiles; y = 8 row chunks of 1024 rows
    lfq_entropy<<<dim3(256, 8), dim3(128), 0, stream>>>(apack, rowc, colsum, scal);
    lfq_colentropy<<<dim3(CS / 256), dim3(256), 0, stream>>>(colsum, scal);
    lfq_finalize<<<dim3(1), dim3(1), 0, stream>>>(scal, auxout);
}
